// RetrievalTool_24172075942189
// MI455X (gfx1250) — compile-verified, matrix operand order checked by reading, not run on hardware
//
#include <hip/hip_runtime.h>
#include <hip/hip_bf16.h>
#include <math.h>

// ---------------- constants ----------------
#define PP 3
#define SEQ 192
#define PRED 96
#define CH 8
#define FDIM 1536            // SEQ*CH
#define NTRAIN 10000
#define BSZ 32
#define TOPM 20
#define DMDIM 512
#define MASK_HALF 287        // SEQ+PRED-1

#define AX_M   (PP * NTRAIN)   // 30000 real rows
#define AX_MP  30080           // padded to multiple of 128
#define AX_PAD (AX_MP - AX_M)  // 80 zero rows

// axnorm macro-tile geometry
#define BM 128
#define BN 64
#define KC 64
#define LDR 144   // LDS row stride in bytes (64 elems * 2B + 16B TDM pad) -> conflict-free & 16B aligned

typedef __bf16 v16bf __attribute__((ext_vector_type(16)));
typedef __bf16 v8bf  __attribute__((ext_vector_type(8)));
typedef float  v8f   __attribute__((ext_vector_type(8)));
typedef unsigned int  u32x4 __attribute__((ext_vector_type(4)));
typedef int           i32x8 __attribute__((ext_vector_type(8)));
typedef int           i32x4 __attribute__((ext_vector_type(4)));

#if __has_builtin(__builtin_amdgcn_tensor_load_to_lds)
#define USE_TDM 1
#else
#define USE_TDM 0
#endif

__device__ __forceinline__ void wait_tensorcnt0() {
#if __has_builtin(__builtin_amdgcn_s_wait_tensorcnt)
    __builtin_amdgcn_s_wait_tensorcnt((short)0);
#else
    asm volatile("s_wait_tensorcnt 0x0" ::: "memory");
#endif
}

#if USE_TDM
// Issue one 2D TDM tile load: global (row-major, stride0 elems per row) -> LDS with padding.
// data_size = 2 bytes (bf16). pad: +16B after every 128B => LDS row stride = tile0*2 + pads.
__device__ __forceinline__ void tdm_load_2d(unsigned ldsAddr, const void* gptr,
                                            unsigned tdim0, unsigned tdim1,
                                            unsigned tile0, unsigned tile1,
                                            unsigned stride0) {
    unsigned long long ga = (unsigned long long)gptr;
    u32x4 g0;
    g0[0] = 1u;                                   // count=1, user descriptor
    g0[1] = ldsAddr;                              // LDS byte address
    g0[2] = (unsigned)(ga & 0xffffffffu);         // global_addr[31:0]
    g0[3] = (unsigned)((ga >> 32) & 0x1ffffffu) | (2u << 30);  // addr[56:32] | type=2
    i32x8 g1;
    g1[0] = (int)((1u << 16)    // data_size = 1 -> 2 bytes
                  | (1u << 20)  // pad_enable
                  | (4u << 22)  // pad_interval code 4 -> every 32 DWORDs (128B)
                  | (3u << 25)  // pad_amount  code 3 -> 4 DWORDs (16B)
                  );
    g1[1] = (int)((tdim0 & 0xffffu) << 16);                       // tensor_dim0[15:0]
    g1[2] = (int)((tdim0 >> 16) | ((tdim1 & 0xffffu) << 16));     // td0[31:16] | td1[15:0]
    g1[3] = (int)((tdim1 >> 16) | (tile0 << 16));                 // td1[31:16] | tile_dim0
    g1[4] = (int)(tile1);                                         // tile_dim1 (tile_dim2=0)
    g1[5] = (int)stride0;                                         // dim0 stride lo32
    g1[6] = 0;                                                    // stride hi16 | stride1 lo
    g1[7] = 0;
    i32x4 z4 = {0, 0, 0, 0};
    i32x8 z8 = {0, 0, 0, 0, 0, 0, 0, 0};
    __builtin_amdgcn_tensor_load_to_lds(g0, g1, z4, z4, z8, 0);
}
#endif

// Global-memory fragment load, CDNA5 16-bit operand layout:
// lanes 0-15 hold K = {k0+0..7, k0+16..23}, lanes 16-31 hold K = {k0+8..15, k0+24..31}.
__device__ __forceinline__ v16bf load_frag(const __hip_bfloat16* __restrict__ rowbase,
                                           int k0, int half) {
    const v8bf lo = *reinterpret_cast<const v8bf*>((const void*)(rowbase + k0 + half * 8));
    const v8bf hi = *reinterpret_cast<const v8bf*>((const void*)(rowbase + k0 + 16 + half * 8));
    v16bf r;
#pragma unroll
    for (int i = 0; i < 8; ++i) { r[i] = lo[i]; r[i + 8] = hi[i]; }
    return r;
}

// LDS fragment load from a padded tile row (rowbase points at row start, kb = ks*64 + half*16)
__device__ __forceinline__ v16bf lds_frag(const char* rowbase, int kb) {
    const v8bf lo = *reinterpret_cast<const v8bf*>(rowbase + kb);
    const v8bf hi = *reinterpret_cast<const v8bf*>(rowbase + kb + 32);
    v16bf r;
#pragma unroll
    for (int i = 0; i < 8; ++i) { r[i] = lo[i]; r[i + 8] = hi[i]; }
    return r;
}

__device__ __forceinline__ int period_of(int p) { return (p == 0) ? 4 : ((p == 1) ? 2 : 1); }

// ---------------- utility kernels ----------------
__global__ void k_zero_f32(float* __restrict__ p, int n) {
    int id = blockIdx.x * 256 + threadIdx.x;
    if (id < n) p[id] = 0.f;
}

__global__ void k_zero_bf16(__hip_bfloat16* __restrict__ p, int n) {
    int id = blockIdx.x * 256 + threadIdx.x;
    if (id < n) p[id] = __float2bfloat16(0.f);
}

// Multi-granularity decomposition: out[p][n][s*CH+c] = groupmean(s) - groupmean(last)
__global__ void k_decomp(const float* __restrict__ in, __hip_bfloat16* __restrict__ obf,
                         float* __restrict__ of32, int N, int L) {
    int id = blockIdx.x * 256 + threadIdx.x;
    int total = PP * N * CH;
    if (id >= total) return;
    int c = id & 7;
    int n = (id >> 3) % N;
    int p = id / (N * CH);
    int g = period_of(p);
    const float* row = in + (size_t)n * L * CH + c;
    float lm = 0.f;
    for (int j = L - g; j < L; ++j) lm += row[j * CH];
    lm /= (float)g;
    size_t ob = ((size_t)p * N + n) * (size_t)(L * CH) + c;
    for (int s0 = 0; s0 < L; s0 += g) {
        float mn = 0.f;
        for (int j = s0; j < s0 + g; ++j) mn += row[j * CH];
        mn /= (float)g;
        float v = mn - lm;
        for (int s = s0; s < s0 + g; ++s) {
            obf[ob + (size_t)s * CH] = __float2bfloat16(v);
            if (of32) of32[ob + (size_t)s * CH] = v;
        }
    }
}

// W_sim -> bf16 and transposed bf16
__global__ void k_conv_w(const float* __restrict__ W, __hip_bfloat16* __restrict__ Wbf,
                         __hip_bfloat16* __restrict__ WTbf) {
    int id = blockIdx.x * 256 + threadIdx.x;
    if (id >= FDIM * FDIM) return;
    int i = id / FDIM, j = id % FDIM;
    float w = W[id];
    Wbf[id] = __float2bfloat16(w);
    WTbf[(size_t)j * FDIM + i] = __float2bfloat16(w);
}

// Rank-8 attention precompute: G = Wq Wk^T (8x8), wqbk = Wq bk, wkbq = Wk bq, c0 = bq.bk
__global__ void k_prep(const float* __restrict__ Wq, const float* __restrict__ bq,
                       const float* __restrict__ Wk, const float* __restrict__ bk,
                       float* __restrict__ Gbuf) {
    int tid = threadIdx.x;
    if (tid < 64) {
        int i = tid >> 3, j = tid & 7;
        float s = 0.f;
        for (int d = 0; d < DMDIM; ++d) s += Wq[i * DMDIM + d] * Wk[j * DMDIM + d];
        Gbuf[tid] = s;
    } else if (tid < 72) {
        int i = tid - 64;
        float s = 0.f;
        for (int d = 0; d < DMDIM; ++d) s += Wq[i * DMDIM + d] * bk[d];
        Gbuf[64 + i] = s;
    } else if (tid < 80) {
        int j = tid - 72;
        float s = 0.f;
        for (int d = 0; d < DMDIM; ++d) s += Wk[j * DMDIM + d] * bq[d];
        Gbuf[72 + j] = s;
    } else if (tid == 80) {
        float s = 0.f;
        for (int d = 0; d < DMDIM; ++d) s += bq[d] * bk[d];
        Gbuf[80] = s;
    }
}

// ---------------- small direct WMMA GEMM (one tile per wave; M=96 jobs) ----------------
__global__ void k_gemm_wmma(const __hip_bfloat16* __restrict__ A,
                            const __hip_bfloat16* __restrict__ BT,
                            const float* __restrict__ bias,
                            float* __restrict__ Cf, __hip_bfloat16* __restrict__ Cbf,
                            int M, int N, int K) {
    int wid = blockIdx.x * 8 + (threadIdx.x >> 5);
    int tiles_n = N >> 4, tiles_m = M >> 4;
    if (wid >= tiles_m * tiles_n) return;
    int mt = wid / tiles_n, nt = wid % tiles_n;
    int lane = threadIdx.x & 31;
    int lr = lane & 15, half = lane >> 4;
    const __hip_bfloat16* arow = A + (size_t)(mt * 16 + lr) * K;
    const __hip_bfloat16* brow = BT + (size_t)(nt * 16 + lr) * K;
    v8f acc = {};
    for (int k0 = 0; k0 < K; k0 += 32) {
        __builtin_prefetch((const void*)(brow + k0 + 512), 0, 0);
        v16bf a = load_frag(arow, k0, half);
        v16bf b = load_frag(brow, k0, half);
        acc = __builtin_amdgcn_wmma_f32_16x16x32_bf16(false, a, false, b, (short)0, acc,
                                                      false, false);
    }
    int col = nt * 16 + lr;
    float bv = bias ? bias[col] : 0.f;
#pragma unroll
    for (int v = 0; v < 8; ++v) {
        int row = mt * 16 + v + half * 8;
        float val = acc[v] + bv;
        if (Cf) Cf[(size_t)row * N + col] = val;
        else    Cbf[(size_t)row * N + col] = __float2bfloat16(val);
    }
}

// ---------------- LDS-tiled ||ax||^2 kernel (the 141.6 GFLOP GEMM) ----------------
// Block = 256 threads = 8 waves; macro-tile 128x64; wave (wm,wn) computes a 32x32 sub-block
// (2x2 WMMA tiles). Operands staged per KC=64 chunk into LDS by the Tensor Data Mover
// (tensor_load_to_lds + s_wait_tensorcnt) or by a manual b128 copy fallback.
__global__ void k_axnorm_tiled(const __hip_bfloat16* __restrict__ A,
                               const __hip_bfloat16* __restrict__ BT,
                               const float* __restrict__ bias,
                               float* __restrict__ sumsq) {
    __shared__ __align__(16) char As[BM * LDR];   // 18432 B
    __shared__ __align__(16) char Bs[BN * LDR];   //  9216 B
    const int ntiles = FDIM / BN;                 // 24
    int bid = blockIdx.x;
    int mt = bid / ntiles, nt = bid % ntiles;
    int m0 = mt * BM, n0 = nt * BN;
    int tid = threadIdx.x;
    int wid = tid >> 5, lane = tid & 31;
    int lr = lane & 15, half = lane >> 4;
    int wm = wid >> 1, wn = wid & 1;              // 4 x 2 wave grid
    v8f acc00 = {}, acc01 = {}, acc10 = {}, acc11 = {};

    for (int k0 = 0; k0 < FDIM; k0 += KC) {
#if USE_TDM
        if (wid == 0) {
            // A tile: 128 rows x 64 elems of train_mg (padded to AX_MP rows)
            tdm_load_2d((unsigned)(size_t)(void*)As,
                        (const void*)(A + (size_t)m0 * FDIM + k0),
                        FDIM, AX_MP, KC, BM, FDIM);
            // B tile: 64 rows x 64 elems of W_sim^T
            tdm_load_2d((unsigned)(size_t)(void*)Bs,
                        (const void*)(BT + (size_t)n0 * FDIM + k0),
                        FDIM, FDIM, KC, BN, FDIM);
            wait_tensorcnt0();
        }
#else
        // manual staging: (BM+BN) rows * 8 chunks of 16B = 1536 chunks, 6 per thread
        for (int ci = tid; ci < (BM + BN) * 8; ci += 256) {
            int isB = ci >= BM * 8;
            int li = isB ? ci - BM * 8 : ci;
            int row = li >> 3, seg = li & 7;
            const __hip_bfloat16* src =
                (isB ? BT + (size_t)(n0 + row) * FDIM : A + (size_t)(m0 + row) * FDIM) +
                k0 + seg * 8;
            char* dst = (isB ? Bs : As) + row * LDR + seg * 16;
            *reinterpret_cast<uint4*>(dst) = *reinterpret_cast<const uint4*>((const void*)src);
        }
#endif
        __syncthreads();
#pragma unroll
        for (int ks = 0; ks < 2; ++ks) {
            int kb = ks * 64 + half * 16;
            v16bf a0 = lds_frag(As + (wm * 32 + lr) * LDR, kb);
            v16bf a1 = lds_frag(As + (wm * 32 + 16 + lr) * LDR, kb);
            v16bf b0 = lds_frag(Bs + (wn * 32 + lr) * LDR, kb);
            v16bf b1 = lds_frag(Bs + (wn * 32 + 16 + lr) * LDR, kb);
            acc00 = __builtin_amdgcn_wmma_f32_16x16x32_bf16(false, a0, false, b0, (short)0,
                                                            acc00, false, false);
            acc01 = __builtin_amdgcn_wmma_f32_16x16x32_bf16(false, a0, false, b1, (short)0,
                                                            acc01, false, false);
            acc10 = __builtin_amdgcn_wmma_f32_16x16x32_bf16(false, a1, false, b0, (short)0,
                                                            acc10, false, false);
            acc11 = __builtin_amdgcn_wmma_f32_16x16x32_bf16(false, a1, false, b1, (short)0,
                                                            acc11, false, false);
        }
        __syncthreads();
    }

    // epilogue: add bias, square, reduce over the 16 columns of each sub-tile, accumulate per row
    float bv0 = bias[n0 + wn * 32 + lr];
    float bv1 = bias[n0 + wn * 32 + 16 + lr];
#pragma unroll
    for (int mi = 0; mi < 2; ++mi) {
#pragma unroll
        for (int ni = 0; ni < 2; ++ni) {
            v8f acc = (mi == 0) ? (ni == 0 ? acc00 : acc01) : (ni == 0 ? acc10 : acc11);
            float bv = ni ? bv1 : bv0;
#pragma unroll
            for (int v = 0; v < 8; ++v) {
                float t = acc[v] + bv;
                float s = t * t;
                s += __shfl_xor(s, 1, 32);
                s += __shfl_xor(s, 2, 32);
                s += __shfl_xor(s, 4, 32);
                s += __shfl_xor(s, 8, 32);
                if (lr == 0)
                    atomicAdd(&sumsq[m0 + wm * 32 + mi * 16 + v + half * 8], s);
            }
        }
    }
}

// sim[p,b,n] = (t[p,b]·train_mg[p,n] + ub[p,b]) * inv||ax[p,n]||, window-masked to -inf
__global__ void k_sim_wmma(const __hip_bfloat16* __restrict__ Tb,
                           const __hip_bfloat16* __restrict__ MGb,
                           const float* __restrict__ ub, const float* __restrict__ invn,
                           const int* __restrict__ index, float* __restrict__ sim) {
    const int tiles_b = BSZ / 16, tiles_n = NTRAIN / 16;
    int wid = blockIdx.x * 8 + (threadIdx.x >> 5);
    if (wid >= PP * tiles_b * tiles_n) return;
    int p = wid / (tiles_b * tiles_n);
    int r = wid % (tiles_b * tiles_n);
    int bt = r / tiles_n, nt = r % tiles_n;
    int lane = threadIdx.x & 31;
    int lr = lane & 15, half = lane >> 4;
    const __hip_bfloat16* arow = Tb + (size_t)(p * BSZ + bt * 16 + lr) * FDIM;
    const __hip_bfloat16* brow = MGb + (size_t)(p * NTRAIN + nt * 16 + lr) * FDIM;
    v8f acc = {};
    for (int k0 = 0; k0 < FDIM; k0 += 32) {
        __builtin_prefetch((const void*)(brow + k0 + 512), 0, 0);
        v16bf a = load_frag(arow, k0, half);
        v16bf b = load_frag(brow, k0, half);
        acc = __builtin_amdgcn_wmma_f32_16x16x32_bf16(false, a, false, b, (short)0, acc,
                                                      false, false);
    }
    int n = nt * 16 + lr;
    float iv = invn[p * NTRAIN + n];
#pragma unroll
    for (int v = 0; v < 8; ++v) {
        int b = bt * 16 + v + half * 8;
        int row = p * BSZ + b;
        float val = (acc[v] + ub[row]) * iv;
        int ib = index[b];
        if (n >= ib - MASK_HALF && n <= ib + MASK_HALF) val = -INFINITY;
        sim[(size_t)row * NTRAIN + n] = val;
    }
}

// u = l2norm(bx) rows -> bf16, plus ub = u.b_sim
__global__ void k_row_u(const float* __restrict__ bx, const float* __restrict__ b_sim,
                        __hip_bfloat16* __restrict__ ub16, float* __restrict__ ub) {
    int row = blockIdx.x;
    int tid = threadIdx.x;
    __shared__ float red[256];
    const float* r = bx + (size_t)row * FDIM;
    float ss = 0.f;
    for (int i = tid; i < FDIM; i += 256) { float v = r[i]; ss += v * v; }
    red[tid] = ss;
    __syncthreads();
    for (int ofs = 128; ofs > 0; ofs >>= 1) {
        if (tid < ofs) red[tid] += red[tid + ofs];
        __syncthreads();
    }
    float inv = 1.f / fmaxf(sqrtf(red[0]), 1e-12f);
    __syncthreads();
    float db = 0.f;
    for (int i = tid; i < FDIM; i += 256) {
        float u = r[i] * inv;
        ub16[(size_t)row * FDIM + i] = __float2bfloat16(u);
        db += u * b_sim[i];
    }
    red[tid] = db;
    __syncthreads();
    for (int ofs = 128; ofs > 0; ofs >>= 1) {
        if (tid < ofs) red[tid] += red[tid + ofs];
        __syncthreads();
    }
    if (tid == 0) ub[row] = red[0];
}

__global__ void k_inv_norm(const float* __restrict__ ss, float* __restrict__ inv, int n) {
    int id = blockIdx.x * 256 + threadIdx.x;
    if (id < n) inv[id] = 1.f / fmaxf(sqrtf(ss[id]), 1e-12f);
}

// ---------------- top-20 selection, one block per (p,b) ----------------
__global__ void k_topk(const float* __restrict__ sim, int* __restrict__ topm) {
    int pb = blockIdx.x;
    int tid = threadIdx.x;
    const float* row = sim + (size_t)pb * NTRAIN;
    __shared__ float rv[256];
    __shared__ int   ri[256];
    __shared__ int   sel[TOPM];
    for (int r = 0; r < TOPM; ++r) {
        float best = -INFINITY;
        int bi = 0x7fffffff;
        for (int i = tid; i < NTRAIN; i += 256) {
            bool taken = false;
            for (int j = 0; j < r; ++j) if (sel[j] == i) taken = true;
            if (taken) continue;
            float v = row[i];
            if (v > best || (v == best && i < bi)) { best = v; bi = i; }
        }
        rv[tid] = best; ri[tid] = bi;
        __syncthreads();
        for (int ofs = 128; ofs > 0; ofs >>= 1) {
            if (tid < ofs) {
                if (rv[tid + ofs] > rv[tid] ||
                    (rv[tid + ofs] == rv[tid] && ri[tid + ofs] < ri[tid])) {
                    rv[tid] = rv[tid + ofs]; ri[tid] = ri[tid + ofs];
                }
            }
            __syncthreads();
        }
        if (tid == 0) { sel[r] = ri[0]; topm[pb * TOPM + r] = ri[0]; }
        __syncthreads();
    }
}

// ---------------- fused rank-8 attention + softmax + output, one block per (p,b) ----------------
__global__ void k_final(const float* __restrict__ xmg, const float* __restrict__ train,
                        const float* __restrict__ ydata, const int* __restrict__ topm,
                        const float* __restrict__ Gbuf, float* __restrict__ out) {
    int pb = blockIdx.x;
    int p = pb / BSZ;
    int tid = threadIdx.x;
    int g = period_of(p);
    __shared__ __hip_bfloat16 kf[TOPM * SEQ * CH];   // 61440 B
    __shared__ int   s_idx[TOPM];
    __shared__ float s_nw[TOPM];
    if (tid < TOPM) { s_idx[tid] = topm[pb * TOPM + tid]; s_nw[tid] = 0.f; }
    __syncthreads();
    for (int e = tid; e < TOPM * SEQ * CH; e += 256) {
        int m = e / (SEQ * CH);
        int r = e % (SEQ * CH);
        int t = r >> 3, c = r & 7;
        int idx = s_idx[m];
        const float* trow = train + (size_t)idx * SEQ * CH + c;
        int t0 = (t / g) * g;
        float mn = 0.f;
        for (int j = t0; j < t0 + g; ++j) mn += trow[j * CH];
        mn /= (float)g;
        float lm = 0.f;
        for (int j = SEQ - g; j < SEQ; ++j) lm += trow[j * CH];
        lm /= (float)g;
        kf[e] = __float2bfloat16(mn - lm);
    }
    __syncthreads();
    const float scale = 0.044194173824159216f;  // 1/sqrt(512)
    if (tid < SEQ) {
        const float* xr = xmg + ((size_t)pb * SEQ + tid) * CH;
        float q[8];
        float bias_s = Gbuf[80];
#pragma unroll
        for (int c = 0; c < 8; ++c) {
            float s = Gbuf[72 + c];
#pragma unroll
            for (int cp = 0; cp < 8; ++cp) s += xr[cp] * Gbuf[cp * 8 + c];
            q[c] = s;
            bias_s += xr[c] * Gbuf[64 + c];
        }
        float mx = -INFINITY;
        for (int m = 0; m < TOPM; ++m) {
            const __hip_bfloat16* kr = &kf[m * SEQ * CH];
            for (int t = 0; t < SEQ; ++t) {
                float s = bias_s;
#pragma unroll
                for (int c = 0; c < 8; ++c) s += q[c] * __bfloat162float(kr[t * 8 + c]);
                s *= scale;
                mx = fmaxf(mx, s);
            }
        }
        float Tm[TOPM];
        float Z = 0.f;
        for (int m = 0; m < TOPM; ++m) {
            const __hip_bfloat16* kr = &kf[m * SEQ * CH];
            float tm = 0.f;
            for (int t = 0; t < SEQ; ++t) {
                float s = bias_s;
#pragma unroll
                for (int c = 0; c < 8; ++c) s += q[c] * __bfloat162float(kr[t * 8 + c]);
                s *= scale;
                tm += __expf(s - mx);
            }
            Tm[m] = tm;
            Z += tm;
        }
        float iZ = 1.f / Z;
        for (int m = 0; m < TOPM; ++m) atomicAdd(&s_nw[m], Tm[m] * iZ);
    }
    __syncthreads();
    if (tid == 0) {
        float mx = -INFINITY;
        for (int m = 0; m < TOPM; ++m) mx = fmaxf(mx, s_nw[m]);
        float z = 0.f;
        for (int m = 0; m < TOPM; ++m) { float e = __expf(s_nw[m] - mx); s_nw[m] = e; z += e; }
        float iz = 1.f / z;
        for (int m = 0; m < TOPM; ++m) s_nw[m] *= iz;
    }
    __syncthreads();
    for (int i = tid; i < PRED * CH; i += 256) {
        int l = i >> 3, c = i & 7;
        float acc = 0.f;
        for (int m = 0; m < TOPM; ++m) {
            int idx = s_idx[m];
            const float* yr = ydata + (size_t)idx * PRED * CH + c;
            int t0 = (l / g) * g;
            float mn = 0.f;
            for (int j = t0; j < t0 + g; ++j) mn += yr[j * CH];
            mn /= (float)g;
            float lm = 0.f;
            for (int j = PRED - g; j < PRED; ++j) lm += yr[j * CH];
            lm /= (float)g;
            acc += s_nw[m] * (mn - lm);
        }
        out[((size_t)pb * PRED + l) * CH + c] = acc;
    }
}

// ---------------- host launch ----------------
extern "C" void kernel_launch(void* const* d_in, const int* in_sizes, int n_in,
                              void* d_out, int out_size, void* d_ws, size_t ws_size,
                              hipStream_t stream) {
    const float* x      = (const float*)d_in[0];
    const int*   index  = (const int*)d_in[1];
    const float* train  = (const float*)d_in[2];
    const float* ydata  = (const float*)d_in[3];
    const float* W_sim  = (const float*)d_in[4];
    const float* b_sim  = (const float*)d_in[5];
    const float* Wq     = (const float*)d_in[6];
    const float* bq     = (const float*)d_in[7];
    const float* Wk     = (const float*)d_in[8];
    const float* bk     = (const float*)d_in[9];
    float* out = (float*)d_out;
    (void)in_sizes; (void)n_in; (void)out_size; (void)ws_size;

    char* w = (char*)d_ws;
    auto carve = [&](size_t bytes) -> char* {
        char* p = w;
        w += (bytes + 255) & ~(size_t)255;
        return p;
    };
    __hip_bfloat16* train_mg_bf = (__hip_bfloat16*)carve((size_t)AX_MP * FDIM * 2);
    __hip_bfloat16* WTbf        = (__hip_bfloat16*)carve((size_t)FDIM * FDIM * 2);
    __hip_bfloat16* Wbf         = (__hip_bfloat16*)carve((size_t)FDIM * FDIM * 2);
    float*          x_mg_f32    = (float*)carve((size_t)PP * BSZ * FDIM * 4);
    __hip_bfloat16* x_mg_bf     = (__hip_bfloat16*)carve((size_t)PP * BSZ * FDIM * 2);
    float*          bx          = (float*)carve((size_t)PP * BSZ * FDIM * 4);
    __hip_bfloat16* u_bf        = (__hip_bfloat16*)carve((size_t)PP * BSZ * FDIM * 2);
    float*          ub          = (float*)carve((size_t)PP * BSZ * 4);
    __hip_bfloat16* t_bf        = (__hip_bfloat16*)carve((size_t)PP * BSZ * FDIM * 2);
    float*          sumsq       = (float*)carve((size_t)AX_MP * 4);
    float*          inv_axn     = (float*)carve((size_t)AX_M * 4);
    float*          sim         = (float*)carve((size_t)PP * BSZ * NTRAIN * 4);
    int*            topm        = (int*)carve((size_t)PP * BSZ * TOPM * 4);
    float*          Gbuf        = (float*)carve(96 * 4);

    const int T = 256;
    // 0. zero accumulators and the M-padding rows (ws is poisoned)
    k_zero_f32<<<(AX_MP + T - 1) / T, T, 0, stream>>>(sumsq, AX_MP);
    k_zero_bf16<<<(AX_PAD * FDIM + T - 1) / T, T, 0, stream>>>(
        train_mg_bf + (size_t)AX_M * FDIM, AX_PAD * FDIM);
    // 1. decomposition
    k_decomp<<<(PP * NTRAIN * CH + T - 1) / T, T, 0, stream>>>(train, train_mg_bf, nullptr,
                                                              NTRAIN, SEQ);
    k_decomp<<<(PP * BSZ * CH + T - 1) / T, T, 0, stream>>>(x, x_mg_bf, x_mg_f32, BSZ, SEQ);
    // 2. weight conversions / rank-8 precompute
    k_conv_w<<<(FDIM * FDIM + T - 1) / T, T, 0, stream>>>(W_sim, Wbf, WTbf);
    k_prep<<<1, 128, 0, stream>>>(Wq, bq, Wk, bk, Gbuf);
    // 3. bx = x_mg @ W_sim + b  (WMMA, M=96)
    {
        int tiles = (PP * BSZ / 16) * (FDIM / 16);
        k_gemm_wmma<<<(tiles + 7) / 8, T, 0, stream>>>(x_mg_bf, WTbf, b_sim, bx, nullptr,
                                                       PP * BSZ, FDIM, FDIM);
    }
    // 4. ||ax||^2: LDS-tiled WMMA GEMM with TDM staging (M=30080, N=K=1536)
    {
        int blocks = (AX_MP / BM) * (FDIM / BN);   // 235 * 24
        k_axnorm_tiled<<<blocks, T, 0, stream>>>(train_mg_bf, WTbf, b_sim, sumsq);
    }
    // 5. u = l2norm(bx), ub = u.b_sim ; inverse ax norms
    k_row_u<<<PP * BSZ, T, 0, stream>>>(bx, b_sim, u_bf, ub);
    k_inv_norm<<<(AX_M + T - 1) / T, T, 0, stream>>>(sumsq, inv_axn, AX_M);
    // 6. t = u @ W_sim^T  (WMMA, output bf16)
    {
        int tiles = (PP * BSZ / 16) * (FDIM / 16);
        k_gemm_wmma<<<(tiles + 7) / 8, T, 0, stream>>>(u_bf, Wbf, nullptr, nullptr, t_bf,
                                                       PP * BSZ, FDIM, FDIM);
    }
    // 7. sim = (t . train_mg + ub) * inv||ax||, masked  (WMMA, N=10000)
    {
        int tiles = PP * (BSZ / 16) * (NTRAIN / 16);
        k_sim_wmma<<<(tiles + 7) / 8, T, 0, stream>>>(t_bf, train_mg_bf, ub, inv_axn, index,
                                                      sim);
    }
    // 8. top-20 per (p,b)
    k_topk<<<PP * BSZ, T, 0, stream>>>(sim, topm);
    // 9. fused rank-8 attention + softmax + weighted y output
    k_final<<<PP * BSZ, T, 0, stream>>>(x_mg_f32, train, ydata, topm, Gbuf, out);
}
